// MambaEncoder_61418032332931
// MI455X (gfx1250) — matrix operand order repaired and checked by reading, not compiled, not run on hardware
//
#include <hip/hip_runtime.h>
#include <math.h>

#define TPB 256
static inline int cdiv(int a, int b) { return (a + b - 1) / b; }

typedef __attribute__((ext_vector_type(16))) __bf16 v16bf;
typedef __attribute__((ext_vector_type(8)))  __bf16 v8bf;
typedef __attribute__((ext_vector_type(4)))  __bf16 v4bf;
typedef __attribute__((ext_vector_type(8)))  float  v8f;

enum { ACT_NONE = 0, ACT_SOFTPLUS = 1, ACT_GELU = 2 };

// ---------------------------------------------------------------------------
// Pack f32 [rows,cols](ld) -> zero-padded row-major bf16 [rowsP, colsP]
// (colsP multiple of 32, rowsP multiple of 16); 4 bf16 (8B) per thread.
// ---------------------------------------------------------------------------
__global__ void pack_bf16_kernel(const float* __restrict__ src, int rows, int cols, int ld,
                                 __bf16* __restrict__ dst, int rowsP, int colsP) {
  int cq = colsP >> 2;
  int total = rowsP * cq;
  int idx = blockIdx.x * blockDim.x + threadIdx.x;
  if (idx >= total) return;
  int r = idx / cq, c4 = (idx - r * cq) << 2;
  v4bf v;
#pragma unroll
  for (int j = 0; j < 4; ++j) {
    int c = c4 + j;
    float f = (r < rows && c < cols) ? src[(size_t)r * ld + c] : 0.f;
    v[j] = (__bf16)f;
  }
  *(v4bf*)(dst + (size_t)r * colsP + c4) = v;
}

// ---------------------------------------------------------------------------
// WMMA GEMM on packed bf16:  C[M,N] = act(A[Mp,Kp] @ W[Np,Kp]^T + bias)(+addsrc)
// one wave32 per 16x16 C tile; per 32-K step: 4x global_load_b128 + 1 wmma.
// Fragment layouts per CDNA5 ISA 7.12.2 (row-major sources are contiguous):
//   A: lane m (0-15|16-31=half), chunks at k0+half*8 and k0+16+half*8
//   B: lane n, chunks at k0+half*8 and k0+16+half*8 (same as A)
//   C/D: vgpr r -> M=r+half*8, N=lane&15
// ---------------------------------------------------------------------------
__global__ void wmma_gemm_kernel(const __bf16* __restrict__ Abf, int Kp,
                                 const __bf16* __restrict__ Wbf,
                                 const float* __restrict__ bias,
                                 const float* __restrict__ addsrc,
                                 float* __restrict__ C,
                                 int M, int N, int act) {
  int gwave  = (blockIdx.x * blockDim.x + threadIdx.x) >> 5;
  int lane   = threadIdx.x & 31;
  int tilesN = (N + 15) >> 4;
  int ntiles = ((M + 15) >> 4) * tilesN;
  if (gwave >= ntiles) return;  // wave-uniform: EXEC stays all-ones for WMMA
  int tm = gwave / tilesN, tn = gwave - tm * tilesN;
  int half = lane >> 4, l16 = lane & 15;
  const v8bf* Arow = (const v8bf*)(Abf + (size_t)(tm * 16 + l16) * Kp);
  const v8bf* Wrow = (const v8bf*)(Wbf + (size_t)(tn * 16 + l16) * Kp);
  v8f acc = {0.f, 0.f, 0.f, 0.f, 0.f, 0.f, 0.f, 0.f};
  int ksteps = Kp >> 5;
  for (int ks = 0; ks < ksteps; ++ks) {
    int base = ks << 2;  // v8bf units (8 bf16 each)
    v8bf a0 = Arow[base + half];
    v8bf a1 = Arow[base + 2 + half];
    v8bf b0 = Wrow[base + half];
    v8bf b1 = Wrow[base + 2 + half];
    v16bf a = __builtin_shufflevector(a0, a1, 0, 1, 2, 3, 4, 5, 6, 7,
                                      8, 9, 10, 11, 12, 13, 14, 15);
    v16bf b = __builtin_shufflevector(b0, b1, 0, 1, 2, 3, 4, 5, 6, 7,
                                      8, 9, 10, 11, 12, 13, 14, 15);
    acc = __builtin_amdgcn_wmma_f32_16x16x32_bf16(false, a, false, b,
                                                  (short)0, acc, false, false);
  }
  int col = tn * 16 + l16;
  if (col < N) {
    float bv = bias ? bias[col] : 0.f;
#pragma unroll
    for (int r = 0; r < 8; ++r) {
      int row = tm * 16 + half * 8 + r;
      if (row < M) {
        float v = acc[r] + bv;
        if (act == ACT_SOFTPLUS)      v = (v > 20.f) ? v : log1pf(__expf(v));
        else if (act == ACT_GELU)     v = 0.5f * v * (1.f + erff(v * 0.70710678118f));
        if (addsrc) v += addsrc[(size_t)row * N + col];
        C[(size_t)row * N + col] = v;
      }
    }
  }
}

// ---------------------------------------------------------------------------
// LayerNorm over last dim: one wave per token
// ---------------------------------------------------------------------------
__global__ void layernorm_kernel(const float* __restrict__ x, const float* __restrict__ g,
                                 const float* __restrict__ b, float* __restrict__ y,
                                 int L, int d) {
  int token = blockIdx.x * (blockDim.x >> 5) + (threadIdx.x >> 5);
  int lane  = threadIdx.x & 31;
  if (token >= L) return;
  const float* xr = x + (size_t)token * d;
  float s = 0.f, s2 = 0.f;
  for (int i = lane; i < d; i += 32) { float v = xr[i]; s += v; s2 += v * v; }
#pragma unroll
  for (int o = 16; o; o >>= 1) { s += __shfl_xor(s, o, 32); s2 += __shfl_xor(s2, o, 32); }
  float mean = s / d;
  float var  = s2 / d - mean * mean;
  float inv  = rsqrtf(var + 1e-5f);
  float* yr  = y + (size_t)token * d;
  for (int i = lane; i < d; i += 32) yr[i] = (xr[i] - mean) * inv * g[i] + b[i];
}

// ---------------------------------------------------------------------------
// Patch embed: 4x4x4/s4 conv of (1,1,48,48,48) -> token-major [1728,96]
// ---------------------------------------------------------------------------
__global__ void patch_embed_kernel(const float* __restrict__ x, const float* __restrict__ w,
                                   const float* __restrict__ b, float* __restrict__ out) {
  int idx = blockIdx.x * blockDim.x + threadIdx.x;
  if (idx >= 1728 * 96) return;
  int l = idx / 96, c = idx - 96 * l;
  int h = l / 144, rem = l - h * 144, wv = rem / 12, t = rem - wv * 12;
  const float* wp = w + (size_t)c * 64;
  float acc = b[c];
#pragma unroll
  for (int i = 0; i < 4; ++i)
#pragma unroll
    for (int j = 0; j < 4; ++j)
#pragma unroll
      for (int k = 0; k < 4; ++k)
        acc += x[((size_t)(h * 4 + i) * 48 + (wv * 4 + j)) * 48 + (t * 4 + k)] *
               wp[i * 16 + j * 4 + k];
  out[idx] = acc;
}

// sinusoidal positional embedding added in-place (post-LN), C=96, 12^3 grid
__global__ void posemb_add_kernel(float* __restrict__ xf) {
  int idx = blockIdx.x * blockDim.x + threadIdx.x;
  if (idx >= 1728 * 96) return;
  int l = idx / 96, c = idx - 96 * l;
  int h = l / 144, w = (l / 12) % 12, t = l % 12;
  int axis = c >> 5, cc = c & 31;
  float p = (axis == 0) ? (float)h : (axis == 1) ? (float)w : (float)t;
  float ang = p * __expf(-(float)(cc & 15) * (9.210340371976184f / 16.f));
  xf[idx] += (cc < 16) ? __sinf(ang) : __cosf(ang);
}

// column slice with optional sequence reversal: dst[L,nCols]
__global__ void slice_cols_kernel(const float* __restrict__ src, float* __restrict__ dst,
                                  int L, int srcCols, int colOff, int nCols, int rev) {
  int idx = blockIdx.x * blockDim.x + threadIdx.x;
  if (idx >= L * nCols) return;
  int l = idx / nCols, c = idx - l * nCols;
  int ls = rev ? (L - 1 - l) : l;
  dst[idx] = src[(size_t)ls * srcCols + colOff + c];
}

// depthwise causal conv1d (k=4) + SiLU
__global__ void causal_conv_silu_kernel(const float* __restrict__ xi, const float* __restrict__ cw,
                                        const float* __restrict__ cb, float* __restrict__ xc,
                                        int L, int di) {
  int idx = blockIdx.x * blockDim.x + threadIdx.x;
  if (idx >= L * di) return;
  int l = idx / di, c = idx - l * di;
  float acc = cb[c];
  const float* w = cw + (size_t)c * 4;
#pragma unroll
  for (int j = 0; j < 4; ++j) {
    int ll = l - 3 + j;
    if (ll >= 0) acc += xi[(size_t)ll * di + c] * w[j];
  }
  xc[idx] = acc * (1.f / (1.f + __expf(-acc)));
}

// selective scan: one wave per channel, 2 of 64 states per lane, sequential L
__global__ void selective_scan_kernel(const float* __restrict__ u, const float* __restrict__ dt,
                                      const float* __restrict__ Alog, const float* __restrict__ Dp,
                                      const float* __restrict__ dbl, int dblStride, int drOff,
                                      float* __restrict__ y, int L, int di) {
  int c    = blockIdx.x * (blockDim.x >> 5) + (threadIdx.x >> 5);
  int lane = threadIdx.x & 31;
  if (c >= di) return;
  float A0 = -__expf(Alog[(size_t)c * 64 + lane]);
  float A1 = -__expf(Alog[(size_t)c * 64 + 32 + lane]);
  float Dv = Dp[c];
  float h0 = 0.f, h1 = 0.f;
  for (int l = 0; l < L; ++l) {
    float dtv = dt[(size_t)l * di + c];
    float uv  = u[(size_t)l * di + c];
    const float* bc = dbl + (size_t)l * dblStride + drOff;
    if (l + 1 < L) __builtin_prefetch(dbl + (size_t)(l + 1) * dblStride + drOff, 0, 1);
    float B0 = bc[lane], B1 = bc[32 + lane];
    float C0 = bc[64 + lane], C1 = bc[96 + lane];
    float du = dtv * uv;
    h0 = h0 * __expf(dtv * A0) + du * B0;
    h1 = h1 * __expf(dtv * A1) + du * B1;
    float part = h0 * C0 + h1 * C1;
#pragma unroll
    for (int o = 16; o; o >>= 1) part += __shfl_xor(part, o, 32);
    if (lane == 0) y[(size_t)l * di + c] = part + uv * Dv;
  }
}

// y*silu(z) accumulated into ysum (optionally flipped back for backward dir)
__global__ void gate_accum_kernel(const float* __restrict__ y, const float* __restrict__ z,
                                  float* __restrict__ ysum, int L, int di, int flip, int init) {
  int idx = blockIdx.x * blockDim.x + threadIdx.x;
  if (idx >= L * di) return;
  int l = idx / di, c = idx - l * di;
  float zv = z[idx];
  float g  = y[idx] * zv * (1.f / (1.f + __expf(-zv)));
  int lo = flip ? (L - 1 - l) : l;
  if (init) ysum[(size_t)lo * di + c] = g;
  else      ysum[(size_t)lo * di + c] += g;
}

// FSC: out = concat(cos(pp), sin(pp), gelu_g) + xm  (gelu fused in GEMM)
__global__ void fsc_combine_kernel(const float* __restrict__ pp, const float* __restrict__ g,
                                   const float* __restrict__ xm, float* __restrict__ out,
                                   int L, int d, int p_out) {
  int idx = blockIdx.x * blockDim.x + threadIdx.x;
  if (idx >= L * d) return;
  int l = idx / d, c = idx - l * d;
  float v;
  if (c < p_out)            v = __cosf(pp[(size_t)l * p_out + c]);
  else if (c < 2 * p_out)   v = __sinf(pp[(size_t)l * p_out + (c - p_out)]);
  else                      v = g[(size_t)l * (d - 2 * p_out) + (c - 2 * p_out)];
  out[idx] = v + xm[idx];
}

// patch merge gather: token-major [S^3,C] -> [(S/2)^3, 8C] in offs order
__global__ void merge_gather_kernel(const float* __restrict__ xin, float* __restrict__ xc,
                                    int S, int C) {
  int S2 = S / 2;
  int C8 = 8 * C;
  int total = S2 * S2 * S2 * C8;
  int idx = blockIdx.x * blockDim.x + threadIdx.x;
  if (idx >= total) return;
  const int oi[8] = {0, 1, 0, 0, 1, 0, 1, 1};
  const int oj[8] = {0, 0, 1, 0, 1, 1, 0, 1};
  const int ok[8] = {0, 0, 0, 1, 0, 1, 1, 1};
  int l2 = idx / C8, cc = idx - l2 * C8;
  int q = cc / C, c = cc - q * C;
  int h2 = l2 / (S2 * S2), w2 = (l2 / S2) % S2, t2 = l2 % S2;
  int h = 2 * h2 + oi[q], w = 2 * w2 + oj[q], t = 2 * t2 + ok[q];
  xc[idx] = xin[((size_t)(h * S + w) * S + t) * C + c];
}

// token-major [L,C] -> channel-major output (B,C,H,W,T) flat
__global__ void transpose_out_kernel(const float* __restrict__ a, float* __restrict__ o,
                                     int L, int C) {
  int idx = blockIdx.x * blockDim.x + threadIdx.x;
  if (idx >= L * C) return;
  int l = idx / C, c = idx - l * C;
  o[(size_t)c * L + l] = a[idx];
}

// ===========================================================================
// Host side
// ===========================================================================
struct Arena {
  float* base;
  size_t off;
  float* alloc(size_t n) {
    float* p = base + off;
    off += (n + 63) & ~(size_t)63;
    return p;
  }
};

struct GemmScratch { __bf16* A; __bf16* W; };

struct LayerP {
  const float *norm1_g, *norm1_b, *norm2_g, *norm2_b;
  const float *in_proj_w, *out_proj_w, *p_w, *p_b, *g_w, *g_b;
  const float *conv_w[2], *conv_b[2], *xproj_w[2], *dtproj_w[2], *dtproj_b[2];
  const float *A_log[2], *Dp[2];
};

static void fill_layer(LayerP& lp, void* const* din, int base, bool srt) {
  auto F = [&](int off) { return (const float*)din[base + off]; };
  if (srt) {  // alphabetical pytree order
    lp.A_log[0] = F(0);  lp.A_log[1] = F(1);  lp.Dp[0] = F(2);      lp.Dp[1] = F(3);
    lp.conv_b[0] = F(4); lp.conv_b[1] = F(5); lp.conv_w[0] = F(6);  lp.conv_w[1] = F(7);
    lp.dtproj_b[0] = F(8); lp.dtproj_b[1] = F(9); lp.dtproj_w[0] = F(10); lp.dtproj_w[1] = F(11);
    lp.g_b = F(12); lp.g_w = F(13); lp.in_proj_w = F(14);
    lp.norm1_b = F(15); lp.norm1_g = F(16); lp.norm2_b = F(17); lp.norm2_g = F(18);
    lp.out_proj_w = F(19); lp.p_b = F(20); lp.p_w = F(21);
    lp.xproj_w[0] = F(22); lp.xproj_w[1] = F(23);
  } else {    // dict insertion order
    lp.norm1_g = F(0); lp.norm1_b = F(1); lp.norm2_g = F(2); lp.norm2_b = F(3);
    lp.in_proj_w = F(4); lp.out_proj_w = F(5); lp.p_w = F(6); lp.p_b = F(7);
    lp.g_w = F(8); lp.g_b = F(9);
    lp.conv_w[0] = F(10); lp.conv_b[0] = F(11); lp.xproj_w[0] = F(12);
    lp.dtproj_w[0] = F(13); lp.dtproj_b[0] = F(14); lp.A_log[0] = F(15); lp.Dp[0] = F(16);
    lp.conv_w[1] = F(17); lp.conv_b[1] = F(18); lp.xproj_w[1] = F(19);
    lp.dtproj_w[1] = F(20); lp.dtproj_b[1] = F(21); lp.A_log[1] = F(22); lp.Dp[1] = F(23);
  }
}

static void gemm(const float* A, int lda, const float* W, const float* bias,
                 const float* addsrc, float* C, int M, int N, int K, int act,
                 GemmScratch gs, hipStream_t s) {
  int Kp = (K + 31) & ~31;
  int Mp = (M + 15) & ~15;
  int Np = (N + 15) & ~15;
  pack_bf16_kernel<<<cdiv(Mp * (Kp >> 2), TPB), TPB, 0, s>>>(A, M, K, lda, gs.A, Mp, Kp);
  pack_bf16_kernel<<<cdiv(Np * (Kp >> 2), TPB), TPB, 0, s>>>(W, N, K, K, gs.W, Np, Kp);
  int waves = (Mp >> 4) * (Np >> 4);
  wmma_gemm_kernel<<<cdiv(waves, 8), TPB, 0, s>>>(gs.A, Kp, gs.W, bias, addsrc, C, M, N, act);
}

static void run_mamba_layer(const LayerP& lp, const float* xin, float* xout,
                            int L, int d, Arena ar, GemmScratch gs, hipStream_t s) {
  int di = 4 * d, dr = (d + 15) / 16, xr = dr + 128;
  int p_out = d / 4, g_out = d - 2 * p_out;
  float* ln1  = ar.alloc((size_t)L * d);
  float* xz   = ar.alloc((size_t)L * 2 * di);
  float* xi   = ar.alloc((size_t)L * di);
  float* zf   = ar.alloc((size_t)L * di);
  float* xir  = ar.alloc((size_t)L * di);
  float* zr   = ar.alloc((size_t)L * di);
  float* xc   = ar.alloc((size_t)L * di);
  float* dbl  = ar.alloc((size_t)L * xr);
  float* dt   = ar.alloc((size_t)L * di);
  float* ysc  = ar.alloc((size_t)L * di);
  float* ysum = ar.alloc((size_t)L * di);
  float* xm   = ar.alloc((size_t)L * d);
  float* ln2  = ar.alloc((size_t)L * d);
  float* pp   = ar.alloc((size_t)L * p_out);
  float* gg   = ar.alloc((size_t)L * g_out);

  layernorm_kernel<<<cdiv(L, 8), TPB, 0, s>>>(xin, lp.norm1_g, lp.norm1_b, ln1, L, d);
  gemm(ln1, d, lp.in_proj_w, nullptr, nullptr, xz, L, 2 * di, d, ACT_NONE, gs, s);
  slice_cols_kernel<<<cdiv(L * di, TPB), TPB, 0, s>>>(xz, xi,  L, 2 * di, 0,  di, 0);
  slice_cols_kernel<<<cdiv(L * di, TPB), TPB, 0, s>>>(xz, zf,  L, 2 * di, di, di, 0);
  slice_cols_kernel<<<cdiv(L * di, TPB), TPB, 0, s>>>(xz, xir, L, 2 * di, 0,  di, 1);
  slice_cols_kernel<<<cdiv(L * di, TPB), TPB, 0, s>>>(xz, zr,  L, 2 * di, di, di, 1);

  for (int dir = 0; dir < 2; ++dir) {
    const float* xd = dir ? xir : xi;
    const float* zd = dir ? zr  : zf;
    causal_conv_silu_kernel<<<cdiv(L * di, TPB), TPB, 0, s>>>(xd, lp.conv_w[dir],
                                                              lp.conv_b[dir], xc, L, di);
    gemm(xc, di, lp.xproj_w[dir], nullptr, nullptr, dbl, L, xr, di, ACT_NONE, gs, s);
    gemm(dbl, xr, lp.dtproj_w[dir], lp.dtproj_b[dir], nullptr, dt, L, di, dr,
         ACT_SOFTPLUS, gs, s);
    selective_scan_kernel<<<cdiv(di, 8), TPB, 0, s>>>(xc, dt, lp.A_log[dir], lp.Dp[dir],
                                                      dbl, xr, dr, ysc, L, di);
    gate_accum_kernel<<<cdiv(L * di, TPB), TPB, 0, s>>>(ysc, zd, ysum, L, di, dir,
                                                        dir == 0 ? 1 : 0);
  }

  gemm(ysum, di, lp.out_proj_w, nullptr, xin, xm, L, d, di, ACT_NONE, gs, s);
  layernorm_kernel<<<cdiv(L, 8), TPB, 0, s>>>(xm, lp.norm2_g, lp.norm2_b, ln2, L, d);
  gemm(ln2, d, lp.p_w, lp.p_b, nullptr, pp, L, p_out, d, ACT_NONE, gs, s);
  gemm(ln2, d, lp.g_w, lp.g_b, nullptr, gg, L, g_out, d, ACT_GELU, gs, s);
  fsc_combine_kernel<<<cdiv(L * d, TPB), TPB, 0, s>>>(pp, gg, xm, xout, L, d, p_out);
}

static void run_merge(const float* xin, float* xout, int S, int C,
                      const float* lng, const float* lnb, const float* red,
                      Arena ar, GemmScratch gs, hipStream_t s) {
  int S2 = S / 2, L2 = S2 * S2 * S2, C8 = 8 * C;
  float* xc  = ar.alloc((size_t)L2 * C8);
  float* lnm = ar.alloc((size_t)L2 * C8);
  merge_gather_kernel<<<cdiv(L2 * C8, TPB), TPB, 0, s>>>(xin, xc, S, C);
  layernorm_kernel<<<cdiv(L2, 8), TPB, 0, s>>>(xc, lng, lnb, lnm, L2, C8);
  gemm(lnm, C8, red, nullptr, nullptr, xout, L2, 2 * C, C8, ACT_NONE, gs, s);
}

extern "C" void kernel_launch(void* const* d_in, const int* in_sizes, int n_in,
                              void* d_out, int out_size, void* d_ws, size_t ws_size,
                              hipStream_t stream) {
  (void)out_size; (void)ws_size;
  // Locate x (48^3 = 110592 elements) and infer param flattening convention.
  int xidx = -1;
  for (int i = 0; i < n_in; ++i)
    if (in_sizes[i] == 110592) { xidx = i; break; }
  if (xidx < 0) xidx = n_in - 1;
  int pbase = (xidx == 0) ? 1 : 0;
  bool srt = (in_sizes[pbase] == 768);  // sorted: merge1.ln_b(768); insertion: patch_embed.w(6144)

  const float* x = (const float*)d_in[xidx];
  auto F = [&](int off) { return (const float*)d_in[pbase + off]; };
  const float *pe_w, *pe_b, *pe_lng, *pe_lnb;
  const float *m1_lng, *m1_lnb, *m1_red, *m2_lng, *m2_lnb, *m2_red;
  if (srt) {
    m1_lnb = F(0); m1_lng = F(1); m1_red = F(2);
    m2_lnb = F(3); m2_lng = F(4); m2_red = F(5);
    pe_b = F(6); pe_lnb = F(7); pe_lng = F(8); pe_w = F(9);
  } else {
    pe_w = F(0); pe_b = F(1); pe_lng = F(2); pe_lnb = F(3);
    m1_lng = F(4); m1_lnb = F(5); m1_red = F(6);
    m2_lng = F(7); m2_lnb = F(8); m2_red = F(9);
  }
  LayerP layers[6];
  for (int i = 0; i < 6; ++i) fill_layer(layers[i], d_in, pbase + 10 + 24 * i, srt);

  Arena ar{(float*)d_ws, 0};
  float* bufA = ar.alloc(1728 * 96);
  float* bufB = ar.alloc(1728 * 96);
  float* raw  = ar.alloc(1728 * 96);
  // persistent GEMM packing scratch (max A: 1728x384 bf16; max W: 3072x384 bf16)
  GemmScratch gs;
  gs.A = (__bf16*)ar.alloc((size_t)1728 * 384 / 2 + 64);
  gs.W = (__bf16*)ar.alloc((size_t)3072 * 384 / 2 + 64);
  size_t mark = ar.off;
  float* out = (float*)d_out;

  // patch embed -> LN -> +posemb   (token-major [1728,96])
  patch_embed_kernel<<<cdiv(1728 * 96, TPB), TPB, 0, stream>>>(x, pe_w, pe_b, raw);
  layernorm_kernel<<<cdiv(1728, 8), TPB, 0, stream>>>(raw, pe_lng, pe_lnb, bufA, 1728, 96);
  posemb_add_kernel<<<cdiv(1728 * 96, TPB), TPB, 0, stream>>>(bufA);

  float* cur = bufA;
  float* nxt = bufB;
  // stage 0 (L=1728, d=96)
  for (int l = 0; l < 2; ++l) {
    Arena a2{ar.base, mark};
    run_mamba_layer(layers[l], cur, nxt, 1728, 96, a2, gs, stream);
    float* t = cur; cur = nxt; nxt = t;
  }
  transpose_out_kernel<<<cdiv(1728 * 96, TPB), TPB, 0, stream>>>(cur, out, 1728, 96);

  { Arena a2{ar.base, mark};
    run_merge(cur, nxt, 12, 96, m1_lng, m1_lnb, m1_red, a2, gs, stream);
    float* t = cur; cur = nxt; nxt = t; }

  // stage 1 (L=216, d=192)
  for (int l = 2; l < 4; ++l) {
    Arena a2{ar.base, mark};
    run_mamba_layer(layers[l], cur, nxt, 216, 192, a2, gs, stream);
    float* t = cur; cur = nxt; nxt = t;
  }
  transpose_out_kernel<<<cdiv(216 * 192, TPB), TPB, 0, stream>>>(cur, out + 165888, 216, 192);

  { Arena a2{ar.base, mark};
    run_merge(cur, nxt, 6, 192, m2_lng, m2_lnb, m2_red, a2, gs, stream);
    float* t = cur; cur = nxt; nxt = t; }

  // stage 2 (L=27, d=384)
  for (int l = 4; l < 6; ++l) {
    Arena a2{ar.base, mark};
    run_mamba_layer(layers[l], cur, nxt, 27, 384, a2, gs, stream);
    float* t = cur; cur = nxt; nxt = t;
  }
  transpose_out_kernel<<<cdiv(27 * 384, TPB), TPB, 0, stream>>>(cur, out + 207360, 27, 384);
}
